// CrossDiffusionIMEX2D_79199196938644
// MI455X (gfx1250) — matrix operand order repaired and checked by reading, not compiled, not run on hardware
//
#include <hip/hip_runtime.h>
#include <hip/hip_fp16.h>

// ---------------- problem constants (match reference) ----------------
#define B_    8
#define K_    4
#define H_    256
#define W_    256
#define HID_  16
#define DT_   0.1f
#define STEPS_ 12
#define JITERS_ 8
#define OMEGA_ 0.9f
#define DMAX_ 2.0f
#define CHIMAX_ 2.0f
#define KMAX_ 1.0f

#define KPAD1_ 64     // conv1 im2col K (36) padded to 2x32
#define KPAD2_ 160    // conv2 im2col K (144) padded to 5x32

typedef __attribute__((ext_vector_type(16))) _Float16 v16h;
typedef __attribute__((ext_vector_type(8)))  float    v8f;

#if __has_builtin(__builtin_amdgcn_global_load_async_to_lds_b32) && \
    __has_builtin(__builtin_amdgcn_s_wait_asynccnt)
#define USE_ASYNC_LDS 1
#else
#define USE_ASYNC_LDS 0
#endif

__device__ __forceinline__ float nan0(float v) {
    if (v != v) return 0.f;                       // NaN -> 0
    if (v >  3.4028234e38f) return  3.4028235e38f; // +inf -> fmax
    if (v < -3.4028234e38f) return -3.4028235e38f; // -inf -> -fmax
    return v;
}

__device__ __forceinline__ int clampi(int v, int lo, int hi) {
    return v < lo ? lo : (v > hi ? hi : v);
}

// ---------------- scalar parameter prep ----------------
// prm[0..3]=D_all, prm[4..7]=diag=1+DT*D*4, prm[8]=chi, prm[9]=cap_t
__global__ void cd_param_kernel(const float* __restrict__ D_raw,
                                const float* __restrict__ chi_raw,
                                const float* __restrict__ cap_logit,
                                float* __restrict__ prm) {
    if (blockIdx.x == 0 && threadIdx.x == 0) {
        for (int c = 0; c < K_; ++c) {
            float x  = D_raw[c];
            float sp = (x > 20.f) ? x : log1pf(expf(x));
            float D  = fminf(sp + 1e-8f, DMAX_);
            prm[c]     = D;
            prm[4 + c] = 1.f + DT_ * D * 4.f;
        }
        float x  = chi_raw[0];
        float sp = (x > 20.f) ? x : log1pf(expf(x));
        prm[8] = fminf(sp + 1e-8f, CHIMAX_);
        float s = 1.f / (1.f + expf(-cap_logit[0]));
        prm[9] = fminf(fmaxf(s, 0.001f), 0.95f);
    }
}

// ---------------- weight repack: zero-padded f16 im2col fragments ----------------
// wp1[n*KPAD1_ + k] (k<36 real), wp2[n*KPAD2_ + k] (k<144 real); contiguous in K so a
// lane's B fragment is one aligned 32-byte (v16h) load.
__global__ void cd_wprep(const float* __restrict__ w1, const float* __restrict__ w2,
                         _Float16* __restrict__ wp1, _Float16* __restrict__ wp2) {
    int i = blockIdx.x * blockDim.x + threadIdx.x;
    if (i < HID_ * KPAD1_) {
        int n = i / KPAD1_, kk = i % KPAD1_;
        float v = 0.f;
        if (kk < 36) v = w1[(n * K_ + kk / 9) * 9 + kk % 9];
        wp1[i] = (_Float16)v;
    }
    if (i < HID_ * KPAD2_) {
        int n = i / KPAD2_, kk = i % KPAD2_;
        float v = 0.f;
        if (kk < 144) v = w2[(n * HID_ + kk / 9) * 9 + kk % 9];
        wp2[i] = (_Float16)v;
    }
}

// ---------------- conv1 (K=4 -> HID=16, 3x3, zero pad) via WMMA ----------------
// One wave -> one 16-pixel (row segment) x 16-out-channel tile.
// im2col K-dim = 36, done as 2 chunks of 32 with v_wmma_f32_16x16x32_f16.
__global__ void cd_conv1_wmma(const float* __restrict__ p0,
                              const _Float16* __restrict__ wp1,
                              const float* __restrict__ b1,
                              _Float16* __restrict__ h1) {
    const int tilesPerImg = (H_ * W_) / 16;                 // 4096
    int wave = (blockIdx.x * blockDim.x + threadIdx.x) >> 5;
    int lane = threadIdx.x & 31;
    int b    = wave / tilesPerImg;
    if (b >= B_) return;                                    // wave-uniform
    int rem = wave % tilesPerImg;
    int y   = rem / (W_ / 16);
    int x0  = (rem % (W_ / 16)) * 16;

    int n   = lane & 15;         // out channel (B/C/D column)
    int x   = x0 + (lane & 15);  // pixel (A row)
    int kbA = (lane >> 4) * 8;   // A-layout K-base per half-wave
    int kbB = (lane >> 4) * 16;  // B-layout K-base per half-wave

    v8f c = {};
    float bias = b1[n];
#pragma unroll
    for (int r = 0; r < 8; ++r) c[r] = bias;

#pragma unroll
    for (int kc = 0; kc < 2; ++kc) {
        v16h a;
#pragma unroll
        for (int e = 0; e < 16; ++e) {
            int j = e >> 1, hh = e & 1;
            // A uses contiguous K order: element e -> K=16*(lane/16)+e (matches B)
            int kk = kc * 32 + kbB + e;
            int kkc = kk < 36 ? kk : 35;
            int ci = kkc / 9, tap = kkc % 9;
            int yy = y + tap / 3 - 1, xx = x + tap % 3 - 1;
            bool inb = (kk < 36) & (yy >= 0) & (yy < H_) & (xx >= 0) & (xx < W_);
            float av = p0[((size_t)(b * K_ + ci) * H_ + clampi(yy, 0, H_ - 1)) * W_
                          + clampi(xx, 0, W_ - 1)];
            a[e] = (_Float16)(inb ? av : 0.f);
        }
        v16h bm = *(const v16h*)(wp1 + n * KPAD1_ + kc * 32 + kbB);
        c = __builtin_amdgcn_wmma_f32_16x16x32_f16(false, a, false, bm,
                                                   (short)0, c, false, false);
    }

    int mofs = (lane >> 4) * 8;   // D layout: lanes 16..31 hold M=8..15
#pragma unroll
    for (int r = 0; r < 8; ++r) {
        float v = fmaxf(c[r], 0.f);
        int m = r + mofs;
        h1[((size_t)(b * HID_ + n) * H_ + y) * W_ + (x0 + m)] = (_Float16)v;
    }
}

// ---------------- conv2 (16->16 3x3) + conv3 (16->1 1x1) fused, WMMA ----------------
// K-dim = 144 -> 5 chunks of 32. After WMMA, channels live across lanes; butterfly-
// reduce 16 lanes with w3 folded in, then sigmoid -> k_map.
__global__ void cd_conv23_wmma(const _Float16* __restrict__ h1,
                               const _Float16* __restrict__ wp2,
                               const float* __restrict__ b2,
                               const float* __restrict__ w3,
                               const float* __restrict__ b3,
                               float* __restrict__ kmap) {
    const int tilesPerImg = (H_ * W_) / 16;
    int wave = (blockIdx.x * blockDim.x + threadIdx.x) >> 5;
    int lane = threadIdx.x & 31;
    int b    = wave / tilesPerImg;
    if (b >= B_) return;
    int rem = wave % tilesPerImg;
    int y   = rem / (W_ / 16);
    int x0  = (rem % (W_ / 16)) * 16;

    int n   = lane & 15;
    int x   = x0 + (lane & 15);
    int kbA = (lane >> 4) * 8;
    int kbB = (lane >> 4) * 16;

    v8f c = {};
    float bias = b2[n];
#pragma unroll
    for (int r = 0; r < 8; ++r) c[r] = bias;

#pragma unroll
    for (int kc = 0; kc < 5; ++kc) {
        v16h a;
#pragma unroll
        for (int e = 0; e < 16; ++e) {
            int j = e >> 1, hh = e & 1;
            int kk = kc * 32 + kbB + e;
            int kkc = kk < 144 ? kk : 143;
            int ci = kkc / 9, tap = kkc % 9;
            int yy = y + tap / 3 - 1, xx = x + tap % 3 - 1;
            bool inb = (kk < 144) & (yy >= 0) & (yy < H_) & (xx >= 0) & (xx < W_);
            float av = (float)h1[((size_t)(b * HID_ + ci) * H_ + clampi(yy, 0, H_ - 1)) * W_
                                 + clampi(xx, 0, W_ - 1)];
            a[e] = (_Float16)(inb ? av : 0.f);
        }
        v16h bm = *(const v16h*)(wp2 + n * KPAD2_ + kc * 32 + kbB);
        c = __builtin_amdgcn_wmma_f32_16x16x32_f16(false, a, false, bm,
                                                   (short)0, c, false, false);
    }

    float w3n = w3[n];
    float b3v = b3[0];
    int mofs  = (lane >> 4) * 8;
#pragma unroll
    for (int r = 0; r < 8; ++r) {
        float val = fmaxf(c[r], 0.f) * w3n;     // relu(h2)*w3, per-channel partial
        for (int msk = 1; msk < 16; msk <<= 1)  // sum 16 channels (lanes) per half-wave
            val += __shfl_xor(val, msk, 16);
        if ((lane & 15) == 0) {
            int m = r + mofs;
            float s = 1.f / (1.f + expf(-(val + b3v)));
            kmap[((size_t)b * H_ + y) * W_ + (x0 + m)] = fmaxf(s * KMAX_, 1e-6f);
        }
    }
}

// ---------------- p = max(nan_to_num(p0), 0) ----------------
__global__ void cd_init_p(const float* __restrict__ p0, float* __restrict__ p) {
    int i = blockIdx.x * blockDim.x + threadIdx.x;
    if (i < B_ * K_ * H_ * W_) p[i] = fmaxf(nan0(p0[i]), 0.f);
}

// ---------------- cross-diffusion flux ----------------
__global__ void cd_flux(const float* __restrict__ p, const float* __restrict__ prm,
                        float* __restrict__ fx, float* __restrict__ fy) {
    int i = blockIdx.x * blockDim.x + threadIdx.x;
    if (i >= B_ * H_ * W_) return;
    int x = i % W_, y = (i / W_) % H_, b = i / (H_ * W_);
    int xp = (x < W_ - 1) ? x + 1 : x;
    int yp = (y < H_ - 1) ? y + 1 : y;
    float chi = prm[8];
    float pc[K_], gx[K_], gy[K_], sgx = 0.f, sgy = 0.f;
#pragma unroll
    for (int c = 0; c < K_; ++c) {
        size_t base = (size_t)(b * K_ + c) * H_;
        float v = p[(base + y) * W_ + x];
        gx[c] = nan0(p[(base + y) * W_ + xp] - v);
        gy[c] = nan0(p[(base + yp) * W_ + x] - v);
        pc[c] = v; sgx += gx[c]; sgy += gy[c];
    }
#pragma unroll
    for (int c = 0; c < K_; ++c) {
        size_t idx = ((size_t)(b * K_ + c) * H_ + y) * W_ + x;
        fx[idx] = nan0(-chi * pc[c] * (sgx - gx[c]));
        fy[idx] = nan0(-chi * pc[c] * (sgy - gy[c]));
    }
}

// ---------------- rhs = nan0(p + DT*(div(f) + react)) ----------------
__global__ void cd_rhs(const float* __restrict__ p, const float* __restrict__ fx,
                       const float* __restrict__ fy, const float* __restrict__ kmap,
                       const float* __restrict__ prm, const int* __restrict__ tumor,
                       float* __restrict__ rhs) {
    int i = blockIdx.x * blockDim.x + threadIdx.x;
    if (i >= B_ * K_ * H_ * W_) return;
    int x = i % W_, y = (i / W_) % H_;
    int c = (i / (H_ * W_)) % K_, b = i / (K_ * H_ * W_);
    int xp = (x < W_ - 1) ? x + 1 : x;
    int yp = (y < H_ - 1) ? y + 1 : y;
    size_t base = (size_t)(b * K_ + c) * H_;
    float cross = (fx[i] - fx[(base + y) * W_ + xp]) +
                  (fy[i] - fy[(base + yp) * W_ + x]);
    float react = 0.f;
    if (c == tumor[0]) {
        float pt  = fminf(fmaxf(p[i], 0.f), 1.f);
        float cap = prm[9];
        react = nan0(kmap[((size_t)b * H_ + y) * W_ + x] * pt * (1.f - pt / cap));
    }
    rhs[i] = nan0(p[i] + DT_ * (nan0(cross) + react));
}

// ---------------- Jacobi sweep: xo = nan0(xk + w/diag * (rhs - (xk - DT*D*Lap(xk)))) ----
// 16x16 LDS tile + 18x18 edge-clamped halo; halo loaded via async-to-LDS.
__global__ void cd_jacobi(const float* __restrict__ xk, const float* __restrict__ rhs,
                          const float* __restrict__ prm, float* __restrict__ xo) {
    __shared__ float tile[18][20];
    int s  = blockIdx.z;            // slice = b*K + c
    int c  = s & (K_ - 1);
    int x0 = blockIdx.x * 16, y0 = blockIdx.y * 16;
    const float* src = xk + (size_t)s * H_ * W_;
    int tid = threadIdx.y * 16 + threadIdx.x;

    for (int idx = tid; idx < 18 * 18; idx += 256) {
        int r  = idx / 18, cc = idx % 18;
        int gy = clampi(y0 + r - 1, 0, H_ - 1);
        int gx = clampi(x0 + cc - 1, 0, W_ - 1);
#if USE_ASYNC_LDS
        __builtin_amdgcn_global_load_async_to_lds_b32(
            (__attribute__((address_space(1))) int*)(src + (size_t)gy * W_ + gx),
            (__attribute__((address_space(3))) int*)&tile[r][cc], 0, 0);
#else
        tile[r][cc] = src[(size_t)gy * W_ + gx];
#endif
    }
#if USE_ASYNC_LDS
    __builtin_amdgcn_s_wait_asynccnt(0);
#endif
    __syncthreads();

    int lx = threadIdx.x, ly = threadIdx.y;
    float xc  = tile[ly + 1][lx + 1];
    float lap = tile[ly + 1][lx] + tile[ly + 1][lx + 2] +
                tile[ly][lx + 1] + tile[ly + 2][lx + 1] - 4.f * xc;
    float D = prm[c], dg = prm[4 + c];
    size_t gi = (size_t)s * H_ * W_ + (size_t)(y0 + ly) * W_ + (x0 + lx);
    float r = rhs[gi] - (xc - DT_ * D * lap);
    xo[gi] = nan0(xc + (OMEGA_ / dg) * r);
}

// ---------------- simplex projection over K=4 per pixel ----------------
__global__ void cd_project(const float* __restrict__ xk, float* __restrict__ p) {
    int i = blockIdx.x * blockDim.x + threadIdx.x;
    if (i >= B_ * H_ * W_) return;
    int x = i % W_, y = (i / W_) % H_, b = i / (H_ * W_);

    float v[K_], u[K_];
#pragma unroll
    for (int c = 0; c < K_; ++c) {
        v[c] = nan0(xk[((size_t)(b * K_ + c) * H_ + y) * W_ + x]);
        u[c] = v[c];
    }
    // sort u descending (4-element network)
#define CSWP(a, b) { if (u[a] < u[b]) { float t = u[a]; u[a] = u[b]; u[b] = t; } }
    CSWP(0, 1) CSWP(2, 3) CSWP(0, 2) CSWP(1, 3) CSWP(1, 2)
#undef CSWP
    float css[K_];
    float acc = 0.f;
#pragma unroll
    for (int j = 0; j < K_; ++j) { acc += u[j]; css[j] = acc - 1.f; }
    int rho = 0, bestv = 0;
#pragma unroll
    for (int j = 0; j < K_; ++j) {
        bool cond = (u[j] - css[j] / (float)(j + 1)) > 0.f;
        int val = cond ? (j + 1) : 0;
        if (val > bestv) { bestv = val; rho = j; }
    }
    int ri = rho - 1; if (ri < 0) ri = 0;
    float tau = nan0(css[ri] / (float)(ri + 1));
    float w[K_], sw = 0.f;
#pragma unroll
    for (int c = 0; c < K_; ++c) { w[c] = fmaxf(v[c] - tau, 0.f); sw += w[c]; }
    float inv = 1.f / (sw + 1e-8f);
#pragma unroll
    for (int c = 0; c < K_; ++c)
        p[((size_t)(b * K_ + c) * H_ + y) * W_ + x] = fmaxf(nan0(w[c] * inv), 0.f);
}

// ---------------- host orchestration ----------------
extern "C" void kernel_launch(void* const* d_in, const int* in_sizes, int n_in,
                              void* d_out, int out_size, void* d_ws, size_t ws_size,
                              hipStream_t stream) {
    const float* p0        = (const float*)d_in[0];
    const float* D_raw     = (const float*)d_in[1];
    const float* chi_raw   = (const float*)d_in[2];
    // d_in[3] = k_tumor_raw (computed but unused by the reference forward)
    const float* cap_logit = (const float*)d_in[4];
    const float* w1        = (const float*)d_in[5];
    const float* b1        = (const float*)d_in[6];
    const float* w2        = (const float*)d_in[7];
    const float* b2        = (const float*)d_in[8];
    const float* w3        = (const float*)d_in[9];
    const float* b3        = (const float*)d_in[10];
    const int*   tumor     = (const int*)d_in[11];
    float* out = (float*)d_out;

    const size_t NPIX = (size_t)B_ * H_ * W_;          // 524288
    const size_t NTOT = (size_t)B_ * K_ * H_ * W_;     // 2097152

    char* ws = (char*)d_ws;
    size_t off = 0;
    float*    prm = (float*)(ws + off);    off += 256;
    _Float16* wp1 = (_Float16*)(ws + off); off += 2560;  // 16*64 halves (2KB), 32B-aligned
    _Float16* wp2 = (_Float16*)(ws + off); off += 5632;  // 16*160 halves (5KB)
    float*   kmap = (float*)(ws + off);    off += NPIX * sizeof(float);
    // h1 (f16, 16MB) is reused as fx/fy (2 x f32 fields, 16MB) once convs are done
    _Float16* h1 = (_Float16*)(ws + off);
    float* fx = (float*)(ws + off);
    float* fy = fx + NTOT;
    off += (size_t)B_ * HID_ * H_ * W_ * sizeof(_Float16);
    float* xa = (float*)(ws + off); off += NTOT * sizeof(float);
    float* xb = (float*)(ws + off); off += NTOT * sizeof(float);
    float* rhs = (float*)(ws + off); off += NTOT * sizeof(float);
    (void)ws_size; (void)in_sizes; (void)n_in; (void)out_size;

    cd_param_kernel<<<1, 1, 0, stream>>>(D_raw, chi_raw, cap_logit, prm);
    cd_wprep<<<(HID_ * KPAD2_ + 255) / 256, 256, 0, stream>>>(w1, w2, wp1, wp2);

    const int waves  = (int)(NPIX / 16);   // 32768 tiles, one wave each
    const int blocks = waves / 8;          // 8 waves (256 thr) per block
    cd_conv1_wmma<<<blocks, 256, 0, stream>>>(p0, wp1, b1, h1);
    cd_conv23_wmma<<<blocks, 256, 0, stream>>>(h1, wp2, b2, w3, b3, kmap);

    cd_init_p<<<(int)((NTOT + 255) / 256), 256, 0, stream>>>(p0, out);

    dim3 jgrid(W_ / 16, H_ / 16, B_ * K_);
    dim3 jblk(16, 16, 1);
    for (int s = 0; s < STEPS_; ++s) {
        cd_flux<<<(int)((NPIX + 255) / 256), 256, 0, stream>>>(out, prm, fx, fy);
        cd_rhs<<<(int)((NTOT + 255) / 256), 256, 0, stream>>>(out, fx, fy, kmap,
                                                              prm, tumor, rhs);
        const float* xin = out;
        float* xout = xa;
        for (int j = 0; j < JITERS_; ++j) {
            cd_jacobi<<<jgrid, jblk, 0, stream>>>(xin, rhs, prm, xout);
            xin  = xout;
            xout = (xout == xa) ? xb : xa;
        }
        cd_project<<<(int)((NPIX + 255) / 256), 256, 0, stream>>>(xin, out);
    }
}
